// ForcedDAGPlanner_81063212745675
// MI455X (gfx1250) — matrix operand order repaired and checked by reading, not compiled, not run on hardware
//
#include <hip/hip_runtime.h>
#include <hip/hip_bf16.h>
#include <math.h>

// ---------------------------------------------------------------------------
// ForcedDAGPlanner for MI455X (gfx1250, wave32).
//
// Roofline: naive reference is dominated by K/V projections (275 GFLOP).
// Since q_len==1, both fold into two streaming passes over x (2 x 256 MB ->
// ~22us at 23.3 TB/s HBM). Remaining dense GEMMs (ctx@wo.T, nm@np_w.T) run on
// v_wmma_f32_16x16x32_bf16 with f32 accumulation (16x16 tiles, B=16 = 1 tile).
// top_k: k_nodes = max(3, min(32, 4096//2+1)) = 32 -> node_mask is all ones.
// Streaming passes use b128 loads per lane (4 e-columns/thread) to keep the
// vmem issue rate from capping the 23.3 TB/s stream.
// ---------------------------------------------------------------------------

#define Bv 16
#define Sv 4096
#define Ev 1024
#define Mv 64
#define Nv 32
#define Hv 8
#define Dv 128

typedef __attribute__((ext_vector_type(16))) __bf16 v16bf;
typedef __attribute__((ext_vector_type(8)))  float  v8f;

union BF16x16 { v16bf v; uint4 q[2]; };

// ---- WMMA fragment loaders (layouts per CDNA5 ISA 7.12.2) ------------------
// A (16-bit, 16x32): lanes 0-15 -> M=lane, K base 0/16 ; lanes 16-31 -> K base 8/24.
// Per lane: halves 0..7 = K kb..kb+7 ; halves 8..15 = K kb+16..kb+23.
__device__ __forceinline__ v16bf load_a_bf16(const __bf16* A, int lda, int m0, int k0, int lane) {
  int m  = m0 + (lane & 15);
  int kb = k0 + ((lane >= 16) ? 8 : 0);
  const __bf16* p = A + (size_t)m * lda + kb;
  BF16x16 f;
  f.q[0] = *(const uint4*)(p);
  f.q[1] = *(const uint4*)(p + 16);
  return f.v;
}
// B (16-bit, 32x16), supplied as Bt row-major (N,K): B(k,n) = Bt[n*ldb + k].
// lanes 0-15 -> N=lane, K 0..7,16..23 ; lanes 16-31 -> K 8..15,24..31 (same as A).
__device__ __forceinline__ v16bf load_bt_bf16(const __bf16* Bt, int ldb, int k0, int n0, int lane) {
  int n  = n0 + (lane & 15);
  int kb = k0 + ((lane >= 16) ? 8 : 0);
  const __bf16* p = Bt + (size_t)n * ldb + kb;
  BF16x16 f;
  f.q[0] = *(const uint4*)(p);
  f.q[1] = *(const uint4*)(p + 16);
  return f.v;
}

__device__ __forceinline__ float gelu_exact(float x) {
  return 0.5f * x * (1.0f + erff(x * 0.7071067811865476f));
}
__device__ __forceinline__ float softplus_f(float x) {
  return (x > 20.0f) ? x : log1pf(expf(x));
}
__device__ __forceinline__ float wave_sum(float a) {
  #pragma unroll
  for (int off = 16; off > 0; off >>= 1) a += __shfl_xor(a, off, 32);
  return a;
}

// ---- workspace layout (floats) ---------------------------------------------
#define Q_OFF      0          // 1024
#define U_OFF      1024       // 8*1024
#define QB_OFF     9216       // 8 (padded to 64)
#define ATTN_OFF   9280       // B*H*S = 524288
#define XPART_OFF  533568     // B*8*H*E = 1048576
#define XBAR_OFF   1582144    // B*H*E = 131072
#define CTX_OFF    1713216    // B*E = 16384
#define TR_OFF     1729600    // B*E = 16384
#define NM_OFF     1745984    // B*N*M = 32768
#define F32_WS_END 1778752    // floats -> 7,115,008 bytes
// bf16 region (indices in __bf16 units, based at byte offset F32_WS_END*4)
#define CTXBF_OFF  0          // 16384
#define WOBF_OFF   16384      // 1048576
#define NPWBF_OFF  1064960    // 65536
#define NMBF_OFF   1130496    // 32768
// total ws: ~9.5 MB

// ---- output layout (floats, concatenated in return order) ------------------
#define OUT_NE    0          // (B,N,E) 524288
#define OUT_ADJ   524288     // (B,N,N) 16384
#define OUT_SURF  540672     // (B,)    16
#define OUT_COMP  540688     // (B,N)   512
#define OUT_MASK  541200     // (B,N)   512

// ============================================================================
// K1: convert wo and np_w to bf16
__global__ void __launch_bounds__(256) conv_weights_kernel(const float* wo, const float* npw,
                                                           __bf16* wo_bf, __bf16* npw_bf) {
  int idx = blockIdx.x * 256 + threadIdx.x;
  if (idx < Ev * Ev) {
    wo_bf[idx] = (__bf16)wo[idx];
  } else {
    int j = idx - Ev * Ev;               // j < E*M = 65536
    npw_bf[j] = (__bf16)npw[j];
  }
}

// ============================================================================
// K2: q = tq @ wq.T + bq ; u[h] = q_h @ wk_h (pre-scaled by 1/sqrt(D)) ; qb[h]
__global__ void __launch_bounds__(256) qprep_kernel(const float* tq, const float* wq, const float* bq,
                                                    const float* wk, const float* bk, float* ws) {
  __shared__ float tqL[Ev];
  __shared__ float qL[Ev];
  int t = threadIdx.x;
  for (int e = t; e < Ev; e += 256) tqL[e] = tq[e];
  __syncthreads();
  for (int e = t; e < Ev; e += 256) {
    float a = bq[e];
    const float* row = wq + (size_t)e * Ev;
    for (int j = 0; j < Ev; ++j) a += row[j] * tqL[j];
    qL[e] = a;
    ws[Q_OFF + e] = a;
  }
  __syncthreads();
  const float scale = 0.08838834764831845f;   // 1/sqrt(128)
  for (int e = t; e < Ev; e += 256) {         // u[h][e], coalesced over e
    #pragma unroll
    for (int h = 0; h < Hv; ++h) {
      float a = 0.0f;
      for (int d = 0; d < Dv; ++d) a += qL[h * Dv + d] * wk[(size_t)(h * Dv + d) * Ev + e];
      ws[U_OFF + h * Ev + e] = a * scale;
    }
  }
  if (t < Hv) {
    float a = 0.0f;
    for (int d = 0; d < Dv; ++d) a += qL[t * Dv + d] * bk[t * Dv + d];
    ws[QB_OFF + t] = a * scale;
  }
}

// ============================================================================
// K3: scores[b,h,s] = u[h].x[b,s] + qb[h]   (streaming pass 1 over x)
__global__ void __launch_bounds__(256) scores_kernel(const float* x, const float* ws, float* attn) {
  __shared__ float uL[Hv * Ev];
  int t = threadIdx.x;
  for (int i = t; i < Hv * Ev; i += 256) uL[i] = ws[U_OFF + i];
  __syncthreads();
  int b = blockIdx.y;
  int warp = t >> 5, lane = t & 31;
  float qb[Hv];
  #pragma unroll
  for (int h = 0; h < Hv; ++h) qb[h] = ws[QB_OFF + h];
  for (int r = 0; r < 8; ++r) {
    int s = blockIdx.x * 64 + warp * 8 + r;
    const float* xrow = x + ((size_t)(b * Sv + s)) * Ev;
    float acc[Hv] = {0.f, 0.f, 0.f, 0.f, 0.f, 0.f, 0.f, 0.f};
    #pragma unroll
    for (int it = 0; it < 8; ++it) {
      int e0 = it * 128 + lane * 4;
      float4 xv = *(const float4*)(xrow + e0);
      #pragma unroll
      for (int h = 0; h < Hv; ++h) {
        float4 uv = *(const float4*)(&uL[h * Ev + e0]);
        acc[h] += uv.x * xv.x + uv.y * xv.y + uv.z * xv.z + uv.w * xv.w;
      }
    }
    #pragma unroll
    for (int h = 0; h < Hv; ++h) {
      float a = wave_sum(acc[h]);
      if (lane == 0) attn[(size_t)(b * Hv + h) * Sv + s] = a + qb[h];
    }
  }
}

// ============================================================================
// K4: softmax over S, in place in attn buffer
__global__ void __launch_bounds__(256) softmax_kernel(float* attn) {
  __shared__ float red[256];
  int t = threadIdx.x;
  float* p = attn + (size_t)blockIdx.x * Sv;
  float v[16];
  float mx = -INFINITY;
  #pragma unroll
  for (int i = 0; i < 16; ++i) { v[i] = p[t + 256 * i]; mx = fmaxf(mx, v[i]); }
  red[t] = mx; __syncthreads();
  for (int s = 128; s > 0; s >>= 1) { if (t < s) red[t] = fmaxf(red[t], red[t + s]); __syncthreads(); }
  mx = red[0]; __syncthreads();
  float sum = 0.0f;
  #pragma unroll
  for (int i = 0; i < 16; ++i) { v[i] = expf(v[i] - mx); sum += v[i]; }
  red[t] = sum; __syncthreads();
  for (int s = 128; s > 0; s >>= 1) { if (t < s) red[t] += red[t + s]; __syncthreads(); }
  float inv = 1.0f / red[0];
  #pragma unroll
  for (int i = 0; i < 16; ++i) p[t + 256 * i] = v[i] * inv;
}

// ============================================================================
// K5: xbar partials: xbar[b,h,e] = sum_s attn[b,h,s]*x[b,s,e] (streaming pass 2)
// Each thread owns 4 contiguous e-columns -> global_load_b128 per s-row
// (4x fewer vmem instructions than b32 streaming at identical coalescing).
__global__ void __launch_bounds__(256) xbar_partial_kernel(const float* x, const float* attn, float* part) {
  __shared__ float aL[512 * Hv];
  int t = threadIdx.x;
  int sp = blockIdx.x, b = blockIdx.y;
  for (int idx = t; idx < 512 * Hv; idx += 256) {
    int sl = idx >> 3, h = idx & 7;
    aL[idx] = attn[(size_t)(b * Hv + h) * Sv + sp * 512 + sl];
  }
  __syncthreads();
  int e0 = t * 4;                                  // 256 threads x 4 = 1024 e
  float4 acc[Hv];
  #pragma unroll
  for (int h = 0; h < Hv; ++h) acc[h] = make_float4(0.f, 0.f, 0.f, 0.f);
  const float* xbase = x + ((size_t)(b * Sv + sp * 512)) * Ev + e0;
  for (int sl = 0; sl < 512; ++sl) {
    if (sl + 8 < 512) __builtin_prefetch(xbase + (size_t)(sl + 8) * Ev, 0, 0);
    float4 xv = *(const float4*)(xbase + (size_t)sl * Ev);
    #pragma unroll
    for (int h = 0; h < Hv; ++h) {
      float w = aL[sl * 8 + h];
      acc[h].x += w * xv.x; acc[h].y += w * xv.y;
      acc[h].z += w * xv.z; acc[h].w += w * xv.w;
    }
  }
  #pragma unroll
  for (int h = 0; h < Hv; ++h)
    *(float4*)(&part[(size_t)((b * 8 + sp) * Hv + h) * Ev + e0]) = acc[h];
}

// K6: reduce 8 partials -> xbar (b128 in/out)
__global__ void __launch_bounds__(256) xbar_reduce_kernel(const float* part, float* xbar) {
  int o4 = (blockIdx.x * 256 + threadIdx.x) * 4;   // o4 = (b*8+h)*1024 + e
  int b = o4 >> 13, he = o4 & 8191;
  float4 a = make_float4(0.f, 0.f, 0.f, 0.f);
  #pragma unroll
  for (int sp = 0; sp < 8; ++sp) {
    float4 p = *(const float4*)(&part[(size_t)(b * 8 + sp) * (Hv * Ev) + he]);
    a.x += p.x; a.y += p.y; a.z += p.z; a.w += p.w;
  }
  *(float4*)(&xbar[o4]) = a;
}

// ============================================================================
// K7: ctx[b, h*D+d] = wv[h*D+d] . xbar[b,h] + bv  (one wave per output)
__global__ void __launch_bounds__(256) ctx_kernel(const float* wv, const float* bv, const float* xbar,
                                                  float* ctx, __bf16* ctx_bf) {
  int t = threadIdx.x, lane = t & 31;
  int gw = (blockIdx.x * 256 + t) >> 5;
  int b = gw >> 10, o = gw & 1023, h = o >> 7;
  const float* wrow = wv + (size_t)o * Ev;
  const float* xr   = xbar + (size_t)(b * Hv + h) * Ev;
  float a = 0.0f;
  #pragma unroll
  for (int it = 0; it < 8; ++it) {
    int e0 = it * 128 + lane * 4;
    float4 w4 = *(const float4*)(wrow + e0);
    float4 x4 = *(const float4*)(xr + e0);
    a += w4.x * x4.x + w4.y * x4.y + w4.z * x4.z + w4.w * x4.w;
  }
  a = wave_sum(a);
  if (lane == 0) {
    float v = a + bv[o];
    ctx[b * Ev + o]    = v;
    ctx_bf[b * Ev + o] = (__bf16)v;
  }
}

// ============================================================================
// K8: task_repr(16,1024) = ctx @ wo.T + bo    via v_wmma_f32_16x16x32_bf16
__global__ void __launch_bounds__(256) taskrepr_wmma_kernel(const __bf16* ctx_bf, const __bf16* wo_bf,
                                                            const float* bo, float* tr) {
  int t = threadIdx.x, lane = t & 31;
  int gw = blockIdx.x * 8 + (t >> 5);
  int n0 = gw * 16;
  v8f acc = {};
  for (int kk = 0; kk < 32; ++kk) {
    v16bf a  = load_a_bf16 (ctx_bf, Ev, 0,       kk * 32, lane);
    v16bf bm = load_bt_bf16(wo_bf,  Ev, kk * 32, n0,      lane);
    acc = __builtin_amdgcn_wmma_f32_16x16x32_bf16(false, a, false, bm, (short)0, acc, false, false);
  }
  int n = n0 + (lane & 15);
  int mb = (lane >= 16) ? 8 : 0;
  float bias = bo[n];
  #pragma unroll
  for (int r = 0; r < 8; ++r) tr[(mb + r) * Ev + n] = acc[r] + bias;
}

// ============================================================================
// K9: manifold projection + node generation (per batch)
__global__ void __launch_bounds__(256) manifold_kernel(const float* tr,
    const float* mp_w, const float* mp_b, const float* ln_w, const float* ln_b,
    const float* ng_w1, const float* ng_b1, const float* ng_w2, const float* ng_b2,
    float* nm, __bf16* nm_bf) {
  __shared__ float trL[Ev];
  __shared__ float tmL[Mv];
  __shared__ float hL[2 * Mv];
  __shared__ float nmp[Nv * Mv];
  __shared__ float s_mu, s_rv;
  __shared__ float nmu[Nv], nrv[Nv];
  int b = blockIdx.x, t = threadIdx.x;
  for (int e = t; e < Ev; e += 256) trL[e] = tr[b * Ev + e];
  __syncthreads();
  if (t < Mv) {
    float a = mp_b[t];
    const float* row = mp_w + (size_t)t * Ev;
    for (int e = 0; e < Ev; ++e) a += row[e] * trL[e];
    tmL[t] = a;
  }
  __syncthreads();
  if (t == 0) {
    float mu = 0.0f;
    for (int m = 0; m < Mv; ++m) mu += tmL[m];
    mu *= (1.0f / Mv);
    float var = 0.0f;
    for (int m = 0; m < Mv; ++m) { float d = tmL[m] - mu; var += d * d; }
    var *= (1.0f / Mv);
    s_mu = mu; s_rv = rsqrtf(var + 1e-5f);
  }
  __syncthreads();
  if (t < Mv) tmL[t] = (tmL[t] - s_mu) * s_rv * ln_w[t] + ln_b[t];
  __syncthreads();
  if (t < 2 * Mv) {
    float a = ng_b1[t];
    const float* row = ng_w1 + (size_t)t * Mv;
    for (int m = 0; m < Mv; ++m) a += row[m] * tmL[m];
    hL[t] = gelu_exact(a);
  }
  __syncthreads();
  for (int o = t; o < Nv * Mv; o += 256) {
    float a = ng_b2[o];
    const float* row = ng_w2 + (size_t)o * (2 * Mv);
    for (int j = 0; j < 2 * Mv; ++j) a += row[j] * hL[j];
    nmp[o] = a;
  }
  __syncthreads();
  if (t < Nv) {
    float mu = 0.0f;
    for (int m = 0; m < Mv; ++m) mu += nmp[t * Mv + m];
    mu *= (1.0f / Mv);
    float var = 0.0f;
    for (int m = 0; m < Mv; ++m) { float d = nmp[t * Mv + m] - mu; var += d * d; }
    var *= (1.0f / Mv);
    nmu[t] = mu; nrv[t] = rsqrtf(var + 1e-5f);
  }
  __syncthreads();
  for (int o = t; o < Nv * Mv; o += 256) {
    int i = o >> 6, m = o & 63;
    float v = (nmp[o] - nmu[i]) * nrv[i] * ln_w[m] + ln_b[m];
    nm[(size_t)b * (Nv * Mv) + o]    = v;
    nm_bf[(size_t)b * (Nv * Mv) + o] = (__bf16)v;
  }
}

// ============================================================================
// K10: edges + complexities + surface (node_mask is all-True since k_nodes==N)
__global__ void __launch_bounds__(256) edges_kernel(const float* nm,
    const float* ep_w, const float* ep_b,
    const float* cp_w1, const float* cp_b1, const float* cp_w2, const float* cp_b2,
    float* out) {
  __shared__ float nmL[Nv * Mv];
  __shared__ float tb[Nv * Mv];
  __shared__ float c1[Nv * (Mv / 2)];
  __shared__ float compL[Nv];
  int b = blockIdx.x, t = threadIdx.x;
  for (int o = t; o < Nv * Mv; o += 256) nmL[o] = nm[(size_t)b * (Nv * Mv) + o];
  __syncthreads();
  for (int o = t; o < Nv * Mv; o += 256) {        // t[i][n] = nm[i] @ ep_w[:,n]
    int i = o >> 6, n = o & 63;
    float a = 0.0f;
    for (int m = 0; m < Mv; ++m) a += nmL[i * Mv + m] * ep_w[m * Mv + n];
    tb[o] = a;
  }
  __syncthreads();
  const float thr = -0.8472978603872037f;          // logit(0.3)
  for (int o = t; o < Nv * Nv; o += 256) {
    int i = o >> 5, j = o & 31;
    float l = ep_b[0];
    for (int n = 0; n < Mv; ++n) l += tb[i * Mv + n] * nmL[j * Mv + n];
    out[OUT_ADJ + b * (Nv * Nv) + o] = (j > i && l > thr) ? 1.0f : 0.0f;
  }
  for (int o = t; o < Nv * (Mv / 2); o += 256) {
    int i = o >> 5, p = o & 31;
    float a = cp_b1[p];
    for (int m = 0; m < Mv; ++m) a += nmL[i * Mv + m] * cp_w1[p * Mv + m];
    c1[o] = gelu_exact(a);
  }
  __syncthreads();
  if (t < Nv) {
    float a = cp_b2[0];
    for (int p = 0; p < Mv / 2; ++p) a += c1[t * (Mv / 2) + p] * cp_w2[p];
    float cm = softplus_f(a);
    compL[t] = cm;
    out[OUT_COMP + b * Nv + t] = cm;
    out[OUT_MASK + b * Nv + t] = 1.0f;
  }
  __syncthreads();
  if (t == 0) {
    float s = 0.0f;
    for (int i = 0; i < Nv; ++i) s += compL[i];
    out[OUT_SURF + b] = s;
  }
}

// ============================================================================
// K11: node_embeddings(512,1024) = nm(512,64) @ np_w.T + np_b   via WMMA bf16
__global__ void __launch_bounds__(256) node_emb_wmma_kernel(const __bf16* nm_bf, const __bf16* npw_bf,
                                                            const float* np_b, float* out) {
  int t = threadIdx.x, lane = t & 31;
  int gw = blockIdx.x * 8 + (t >> 5);  // 2048 tiles: 32 m-tiles x 64 n-tiles
  int m0 = (gw >> 6) * 16;
  int n0 = (gw & 63) * 16;
  v8f acc = {};
  #pragma unroll
  for (int kk = 0; kk < 2; ++kk) {
    v16bf a  = load_a_bf16 (nm_bf,  Mv, m0,      kk * 32, lane);
    v16bf bm = load_bt_bf16(npw_bf, Mv, kk * 32, n0,      lane);
    acc = __builtin_amdgcn_wmma_f32_16x16x32_bf16(false, a, false, bm, (short)0, acc, false, false);
  }
  int n = n0 + (lane & 15);
  int mb = m0 + ((lane >= 16) ? 8 : 0);
  float bias = np_b[n];
  #pragma unroll
  for (int r = 0; r < 8; ++r) out[OUT_NE + (size_t)(mb + r) * Ev + n] = acc[r] + bias;
}

// ============================================================================
extern "C" void kernel_launch(void* const* d_in, const int* in_sizes, int n_in,
                              void* d_out, int out_size, void* d_ws, size_t ws_size,
                              hipStream_t stream) {
  (void)in_sizes; (void)n_in; (void)out_size; (void)ws_size;
  const float* x     = (const float*)d_in[0];
  const float* tq    = (const float*)d_in[1];
  const float* wq    = (const float*)d_in[2];
  const float* bq    = (const float*)d_in[3];
  const float* wk    = (const float*)d_in[4];
  const float* bk    = (const float*)d_in[5];
  const float* wv    = (const float*)d_in[6];
  const float* bv    = (const float*)d_in[7];
  const float* wo    = (const float*)d_in[8];
  const float* bo    = (const float*)d_in[9];
  const float* mp_w  = (const float*)d_in[10];
  const float* mp_b  = (const float*)d_in[11];
  const float* ln_w  = (const float*)d_in[12];
  const float* ln_b  = (const float*)d_in[13];
  const float* ng_w1 = (const float*)d_in[14];
  const float* ng_b1 = (const float*)d_in[15];
  const float* ng_w2 = (const float*)d_in[16];
  const float* ng_b2 = (const float*)d_in[17];
  // d_in[18]/[19]: imp_w/imp_b unused — top_k selects all 32 nodes.
  const float* ep_w  = (const float*)d_in[20];
  const float* ep_b  = (const float*)d_in[21];
  const float* cp_w1 = (const float*)d_in[22];
  const float* cp_b1 = (const float*)d_in[23];
  const float* cp_w2 = (const float*)d_in[24];
  const float* cp_b2 = (const float*)d_in[25];
  const float* np_w  = (const float*)d_in[26];
  const float* np_b  = (const float*)d_in[27];

  float*  ws   = (float*)d_ws;
  __bf16* wsbf = (__bf16*)((char*)d_ws + (size_t)F32_WS_END * 4);
  float*  out  = (float*)d_out;

  float*  attn   = ws + ATTN_OFF;
  float*  part   = ws + XPART_OFF;
  float*  xbar   = ws + XBAR_OFF;
  float*  ctx    = ws + CTX_OFF;
  float*  tr     = ws + TR_OFF;
  float*  nm     = ws + NM_OFF;
  __bf16* ctx_bf = wsbf + CTXBF_OFF;
  __bf16* wo_bf  = wsbf + WOBF_OFF;
  __bf16* npw_bf = wsbf + NPWBF_OFF;
  __bf16* nm_bf  = wsbf + NMBF_OFF;

  conv_weights_kernel<<<(Ev * Ev + Ev * Mv) / 256, 256, 0, stream>>>(wo, np_w, wo_bf, npw_bf);
  qprep_kernel<<<1, 256, 0, stream>>>(tq, wq, bq, wk, bk, ws);
  scores_kernel<<<dim3(Sv / 64, Bv), 256, 0, stream>>>(x, ws, attn);
  softmax_kernel<<<Bv * Hv, 256, 0, stream>>>(attn);
  xbar_partial_kernel<<<dim3(8, Bv), 256, 0, stream>>>(x, attn, part);
  xbar_reduce_kernel<<<(Bv * Hv * Ev) / (256 * 4), 256, 0, stream>>>(part, xbar);
  ctx_kernel<<<(Bv * Ev) / 8, 256, 0, stream>>>(wv, bv, xbar, ctx, ctx_bf);
  taskrepr_wmma_kernel<<<(Ev / 16) / 8, 256, 0, stream>>>(ctx_bf, wo_bf, bo, tr);
  manifold_kernel<<<Bv, 256, 0, stream>>>(tr, mp_w, mp_b, ln_w, ln_b,
                                          ng_w1, ng_b1, ng_w2, ng_b2, nm, nm_bf);
  edges_kernel<<<Bv, 256, 0, stream>>>(nm, ep_w, ep_b, cp_w1, cp_b1, cp_w2, cp_b2, out);
  node_emb_wmma_kernel<<<(32 * 64) / 8, 256, 0, stream>>>(nm_bf, npw_bf, np_b, out);
}
